// HiSABlock_58231166599229
// MI455X (gfx1250) — matrix operand order repaired and checked by reading, not compiled, not run on hardware
//
#include <hip/hip_runtime.h>
#include <cstdint>

// ---------------- problem constants (fixed by reference) ----------------
#define D_MODEL 1024
#define NHEAD   16
#define DK      64
#define SEQ     2048
#define BATCH   2
#define DFF     4096
#define MROWS   (BATCH*SEQ)   // 4096 token rows

// ---------------- WMMA types ----------------
typedef __attribute__((ext_vector_type(16))) __bf16 v16bf;
typedef __attribute__((ext_vector_type(8)))  __bf16 v8bf;
typedef __attribute__((ext_vector_type(8)))  float  v8f;

static __device__ __forceinline__ v16bf combine16(v8bf lo, v8bf hi) {
  union { v16bf v; v8bf h[2]; } u;
  u.h[0] = lo; u.h[1] = hi;
  return u.v;
}
static __device__ __forceinline__ v8bf ld8(const __bf16* p) {
  return *(const v8bf*)p;   // 16B load -> global_load_b128 / ds_read_b128
}
static __device__ __forceinline__ v8f wmma_bf16(v16bf a, v16bf b, v8f c) {
  // D = A(16x32 bf16) x B(32x16 bf16) + C(f32), f32 accumulate
  return __builtin_amdgcn_wmma_f32_16x16x32_bf16(
      /*neg_a=*/false, a, /*neg_b=*/false, b,
      /*c_mod=*/(short)0, c, /*reuse_a=*/false, /*reuse_b=*/false);
}

// ---------------- fp32 -> bf16 elementwise ----------------
__global__ void __launch_bounds__(256)
cvt_f32_bf16(const float* __restrict__ in, __bf16* __restrict__ out, int n4) {
  int i = blockIdx.x * 256 + threadIdx.x;
  if (i >= n4) return;
  float4 f = ((const float4*)in)[i];
  union { __bf16 h[4]; uint2 u; } o;
  o.h[0] = (__bf16)f.x; o.h[1] = (__bf16)f.y;
  o.h[2] = (__bf16)f.z; o.h[3] = (__bf16)f.w;
  ((uint2*)out)[i] = o.u;
}

// ---------------- fp32 [R][C] -> bf16 [C][R] (weight transpose) ----------------
__global__ void __launch_bounds__(256)
transpose_cvt(const float* __restrict__ in, __bf16* __restrict__ out, int R, int C) {
  __shared__ float tile[32][33];
  const int c0 = blockIdx.x * 32, r0 = blockIdx.y * 32;
  const int tx = threadIdx.x, ty = threadIdx.y;
#pragma unroll
  for (int i = 0; i < 4; i++)
    tile[ty + i * 8][tx] = in[(size_t)(r0 + ty + i * 8) * C + c0 + tx];
  __syncthreads();
#pragma unroll
  for (int i = 0; i < 4; i++)
    out[(size_t)(c0 + ty + i * 8) * R + r0 + tx] = (__bf16)tile[tx][ty + i * 8];
}

// ---------------- WMMA GEMM: C[M,N] = A[M,K](bf16) * BT[N,K](bf16)^T + bias ---
// 4 waves/block, each wave owns a 64x64 output tile (4x4 sixteens):
// 16 b128 loads -> 16 WMMAs per 32-K step (2x the WMMA density of a 32x32 tile).
// MODE 0: f32 row-major out
// MODE 1: bf16 row-major out, relu (FFN hidden)
// MODE 2: bf16 scatter to q/k layout [b,h,s,d]
// MODE 3: bf16 scatter to v^T layout [b,h,d,s]
template <int MODE>
__global__ void __launch_bounds__(128)
gemm_bf16_wmma(const __bf16* __restrict__ A, const __bf16* __restrict__ BT,
               const float* __restrict__ bias, void* __restrict__ outp,
               int M, int N, int K) {
  const int tid = threadIdx.x;
  const int w   = tid >> 5;          // 4 waves, tiled along N
  const int l   = tid & 31;
  const int lh  = l & 15;
  const int hi1 = l >> 4;
  const int hi8  = hi1 ? 8 : 0;      // A-fragment K split {0..7,16..23}/{8..15,24..31}
  const int hi16 = hi1 ? 16 : 0;     // unused; B is loaded with hi8, same K split as A
  const int wm = blockIdx.y * 64;                  // block M tile = 64
  const int wn = blockIdx.x * 256 + w * 64;        // block N tile = 256

  v8f acc[4][4];
#pragma unroll
  for (int i = 0; i < 4; i++)
#pragma unroll
    for (int j = 0; j < 4; j++) acc[i][j] = (v8f){0,0,0,0,0,0,0,0};

  const __bf16* aBase = A  + (size_t)(wm + lh) * K;
  const __bf16* bBase = BT + (size_t)(wn + lh) * K;

  for (int k0 = 0; k0 < K; k0 += 32) {
    // prefetch next K-slab (global_prefetch_b8)
    if (k0 + 32 < K) {
      __builtin_prefetch(aBase + k0 + 32, 0, 0);
      __builtin_prefetch(bBase + k0 + 32, 0, 0);
    }
    v16bf a[4];
#pragma unroll
    for (int i = 0; i < 4; i++) {
      const __bf16* ap = A + (size_t)(wm + i * 16 + lh) * K + k0 + hi8;
      a[i] = combine16(ld8(ap), ld8(ap + 16));
    }
#pragma unroll
    for (int j = 0; j < 4; j++) {
      const __bf16* bp = BT + (size_t)(wn + j * 16 + lh) * K + k0 + hi8;
      v16bf b = combine16(ld8(bp), ld8(bp + 16));
#pragma unroll
      for (int i = 0; i < 4; i++)
        acc[i][j] = wmma_bf16(a[i], b, acc[i][j]);
    }
  }

#pragma unroll
  for (int i = 0; i < 4; i++) {
#pragma unroll
    for (int j = 0; j < 4; j++) {
      const int nn = wn + j * 16 + lh;
      const float bvl = bias[nn];
#pragma unroll
      for (int r = 0; r < 8; r++) {
        const int mm = wm + i * 16 + r + hi8;         // C layout: M = r + 8*(l>>4)
        float v = acc[i][j][r] + bvl;
        if (MODE == 0) {
          ((float*)outp)[(size_t)mm * N + nn] = v;
        } else if (MODE == 1) {
          ((__bf16*)outp)[(size_t)mm * N + nn] = (__bf16)fmaxf(v, 0.f);
        } else if (MODE == 2) {
          const int b_ = mm >> 11, s_ = mm & (SEQ - 1);
          const int h_ = nn >> 6,  d_ = nn & (DK - 1);
          ((__bf16*)outp)[(((size_t)(b_ * NHEAD + h_)) * SEQ + s_) * DK + d_] = (__bf16)v;
        } else {
          const int b_ = mm >> 11, s_ = mm & (SEQ - 1);
          const int h_ = nn >> 6,  d_ = nn & (DK - 1);
          ((__bf16*)outp)[(((size_t)(b_ * NHEAD + h_)) * DK + d_) * SEQ + s_] = (__bf16)v;
        }
      }
    }
  }
}

// ---------------- flash attention with sparse mask ----------------
// grid (SEQ/64, NHEAD, BATCH), block 128 (4 waves, 16 q-rows each)
__global__ void __launch_bounds__(128)
attn_flash_wmma(const __bf16* __restrict__ qb, const __bf16* __restrict__ kb,
                const __bf16* __restrict__ vt, const unsigned char* __restrict__ mask,
                __bf16* __restrict__ ctxb) {
  constexpr int LP = 72;                         // padded row stride (bf16 elems)
  __shared__ __bf16 ldsP[4 * 16 * LP];           // P staging per wave

  const int tid = threadIdx.x;
  const int w = tid >> 5, l = tid & 31, lh = l & 15;
  const int hi1 = l >> 4;
  const int hi8 = hi1 ? 8 : 0;
  const int hi16 = hi1 ? 16 : 0;
  const int h = blockIdx.y, b = blockIdx.z;
  const int bh = b * NHEAD + h;
  const int qbase = blockIdx.x * 64 + w * 16;
  const float scale = 0.125f;                    // 1/sqrt(64)

  // Q fragments (A operand), loaded once for all k-tiles
  v16bf aq[2];
#pragma unroll
  for (int ks = 0; ks < 2; ks++) {
    const __bf16* qp = qb + ((size_t)bh * SEQ + qbase + lh) * DK + ks * 32 + hi8;
    aq[ks] = combine16(ld8(qp), ld8(qp + 16));
  }

  v8f cacc[4];
#pragma unroll
  for (int j = 0; j < 4; j++) cacc[j] = (v8f){0,0,0,0,0,0,0,0};
  float m_i[8], l_i[8];
#pragma unroll
  for (int r = 0; r < 8; r++) { m_i[r] = -3.0e38f; l_i[r] = 0.f; }

  const unsigned char* mrow = mask + (size_t)bh * SEQ * SEQ;

  for (int kt = 0; kt < SEQ; kt += 64) {
    // ---- scores S = Q K^T  (16 q x 64 k via 8 WMMAs) ----
    v8f sacc[4];
#pragma unroll
    for (int j = 0; j < 4; j++) sacc[j] = (v8f){0,0,0,0,0,0,0,0};
#pragma unroll
    for (int ks = 0; ks < 2; ks++)
#pragma unroll
      for (int j = 0; j < 4; j++) {
        const __bf16* kp = kb + ((size_t)bh * SEQ + kt + j * 16 + lh) * DK + ks * 32 + hi8;
        v16bf bk = combine16(ld8(kp), ld8(kp + 16));
        sacc[j] = wmma_bf16(aq[ks], bk, sacc[j]);
      }
    // ---- mask + scale ----
#pragma unroll
    for (int j = 0; j < 4; j++) {
      const int kc = kt + j * 16 + lh;
#pragma unroll
      for (int r = 0; r < 8; r++) {
        const int q = qbase + r + hi8;
        const unsigned char mv = mrow[(size_t)q * SEQ + kc];
        const float s = sacc[j][r];
        sacc[j][r] = mv ? s * scale : -1.0e30f;
      }
    }
    // ---- online softmax per q-row (rows live across 16-lane halves) ----
#pragma unroll
    for (int r = 0; r < 8; r++) {
      float mx = fmaxf(fmaxf(sacc[0][r], sacc[1][r]), fmaxf(sacc[2][r], sacc[3][r]));
#pragma unroll
      for (int off = 8; off >= 1; off >>= 1) mx = fmaxf(mx, __shfl_xor(mx, off, 32));
      const float mnew = fmaxf(m_i[r], mx);
      const float alpha = __expf(m_i[r] - mnew);
      float rs = 0.f;
#pragma unroll
      for (int j = 0; j < 4; j++) {
        const float p = __expf(sacc[j][r] - mnew);
        rs += p;
        ldsP[w * 16 * LP + (r + hi8) * LP + j * 16 + lh] = (__bf16)p;  // ds_store_b16
      }
#pragma unroll
      for (int off = 8; off >= 1; off >>= 1) rs += __shfl_xor(rs, off, 32);
      l_i[r] = l_i[r] * alpha + rs;
      m_i[r] = mnew;
#pragma unroll
      for (int j = 0; j < 4; j++) cacc[j][r] *= alpha;
    }
    // ---- reload P in A-fragment layout from LDS (same-wave, DS in-order) ----
    v16bf pa[2];
#pragma unroll
    for (int ks = 0; ks < 2; ks++) {
      const __bf16* pp = &ldsP[w * 16 * LP + lh * LP + ks * 32 + hi8];
      pa[ks] = combine16(ld8(pp), ld8(pp + 16));     // ds_read_b128 x2
    }
    // ---- ctx += P * V  (B operand from transposed V) ----
#pragma unroll
    for (int ks = 0; ks < 2; ks++)
#pragma unroll
      for (int j = 0; j < 4; j++) {
        const __bf16* vp = vt + ((size_t)bh * DK + j * 16 + lh) * SEQ + kt + ks * 32 + hi8;
        v16bf bv = combine16(ld8(vp), ld8(vp + 16));
        cacc[j] = wmma_bf16(pa[ks], bv, cacc[j]);
      }
  }

  // ---- normalize and store ctx (bf16, [B*S, D] for the Wo GEMM) ----
#pragma unroll
  for (int j = 0; j < 4; j++) {
    const int d = h * DK + j * 16 + lh;
#pragma unroll
    for (int r = 0; r < 8; r++) {
      const int q = qbase + r + hi8;
      ctxb[((size_t)b * SEQ + q) * D_MODEL + d] = (__bf16)(cacc[j][r] / l_i[r]);
    }
  }
}

// ---------------- residual + LayerNorm (f32 out, optional bf16 out) ---------
__global__ void __launch_bounds__(256)
ln_residual(const float* __restrict__ xa, const float* __restrict__ xr,
            const float* __restrict__ g, const float* __restrict__ beta,
            float* __restrict__ outf, __bf16* __restrict__ outb) {
  __shared__ float wsum[8], wsq[8];
  const int row = blockIdx.x;
  const int tid = threadIdx.x;
  const int w = tid >> 5, l = tid & 31;
  float s[4], sum = 0.f, sq = 0.f;
#pragma unroll
  for (int i = 0; i < 4; i++) {
    const int c = tid + i * 256;
    const float v = xa[(size_t)row * D_MODEL + c] + xr[(size_t)row * D_MODEL + c];
    s[i] = v; sum += v; sq += v * v;
  }
#pragma unroll
  for (int off = 16; off >= 1; off >>= 1) {
    sum += __shfl_xor(sum, off, 32);
    sq  += __shfl_xor(sq,  off, 32);
  }
  if (l == 0) { wsum[w] = sum; wsq[w] = sq; }
  __syncthreads();
  sum = 0.f; sq = 0.f;
#pragma unroll
  for (int i = 0; i < 8; i++) { sum += wsum[i]; sq += wsq[i]; }
  const float mu  = sum * (1.f / D_MODEL);
  const float var = sq * (1.f / D_MODEL) - mu * mu;
  const float rs  = rsqrtf(var + 1e-5f);
#pragma unroll
  for (int i = 0; i < 4; i++) {
    const int c = tid + i * 256;
    const float y = (s[i] - mu) * rs * g[c] + beta[c];
    if (outf) outf[(size_t)row * D_MODEL + c] = y;
    if (outb) outb[(size_t)row * D_MODEL + c] = (__bf16)y;
  }
}

// ---------------- host orchestration ----------------
extern "C" void kernel_launch(void* const* d_in, const int* in_sizes, int n_in,
                              void* d_out, int out_size, void* d_ws, size_t ws_size,
                              hipStream_t stream) {
  (void)in_sizes; (void)n_in; (void)out_size; (void)ws_size;
  const float* x  = (const float*)d_in[0];
  const unsigned char* mask = (const unsigned char*)d_in[1];   // jax bool = 1 byte/elem
  const float* Wq = (const float*)d_in[2];  const float* bq  = (const float*)d_in[3];
  const float* Wk = (const float*)d_in[4];  const float* bk  = (const float*)d_in[5];
  const float* Wv = (const float*)d_in[6];  const float* bv  = (const float*)d_in[7];
  const float* Wo = (const float*)d_in[8];  const float* bo  = (const float*)d_in[9];
  const float* g1 = (const float*)d_in[10]; const float* be1 = (const float*)d_in[11];
  const float* g2 = (const float*)d_in[12]; const float* be2 = (const float*)d_in[13];
  const float* W1 = (const float*)d_in[14]; const float* b1f = (const float*)d_in[15];
  const float* W2 = (const float*)d_in[16]; const float* b2f = (const float*)d_in[17];
  float* out = (float*)d_out;

  char* ws = (char*)d_ws;
  size_t off = 0;
  auto alloc = [&](size_t bytes) -> char* {
    char* p = ws + off;
    off += (bytes + 255) & ~(size_t)255;
    return p;
  };

  __bf16* xb   = (__bf16*)alloc((size_t)MROWS * D_MODEL * 2);   //  8 MB
  __bf16* WqT  = (__bf16*)alloc((size_t)D_MODEL * D_MODEL * 2); //  2 MB
  __bf16* WkT  = (__bf16*)alloc((size_t)D_MODEL * D_MODEL * 2);
  __bf16* WvT  = (__bf16*)alloc((size_t)D_MODEL * D_MODEL * 2);
  __bf16* WoT  = (__bf16*)alloc((size_t)D_MODEL * D_MODEL * 2);
  __bf16* W1T  = (__bf16*)alloc((size_t)D_MODEL * DFF * 2);     //  8 MB
  __bf16* W2T  = (__bf16*)alloc((size_t)DFF * D_MODEL * 2);     //  8 MB
  __bf16* qbuf = (__bf16*)alloc((size_t)MROWS * D_MODEL * 2);   //  8 MB [b,h,s,d]
  __bf16* kbuf = (__bf16*)alloc((size_t)MROWS * D_MODEL * 2);
  __bf16* vtb  = (__bf16*)alloc((size_t)MROWS * D_MODEL * 2);   //  [b,h,d,s]
  __bf16* ctxb = (__bf16*)alloc((size_t)MROWS * D_MODEL * 2);
  float*  aout = (float*)alloc((size_t)MROWS * D_MODEL * 4);    // 16 MB
  float*  y1f  = (float*)alloc((size_t)MROWS * D_MODEL * 4);
  __bf16* y1b  = (__bf16*)alloc((size_t)MROWS * D_MODEL * 2);
  __bf16* hb   = (__bf16*)alloc((size_t)MROWS * DFF * 2);       // 32 MB
  float*  ffw  = (float*)alloc((size_t)MROWS * D_MODEL * 4);

  // 1) precision conversion + weight transposes (bf16)
  cvt_f32_bf16<<<(MROWS * D_MODEL / 4 + 255) / 256, 256, 0, stream>>>(x, xb, MROWS * D_MODEL / 4);
  transpose_cvt<<<dim3(D_MODEL / 32, D_MODEL / 32), dim3(32, 8), 0, stream>>>(Wq, WqT, D_MODEL, D_MODEL);
  transpose_cvt<<<dim3(D_MODEL / 32, D_MODEL / 32), dim3(32, 8), 0, stream>>>(Wk, WkT, D_MODEL, D_MODEL);
  transpose_cvt<<<dim3(D_MODEL / 32, D_MODEL / 32), dim3(32, 8), 0, stream>>>(Wv, WvT, D_MODEL, D_MODEL);
  transpose_cvt<<<dim3(D_MODEL / 32, D_MODEL / 32), dim3(32, 8), 0, stream>>>(Wo, WoT, D_MODEL, D_MODEL);
  transpose_cvt<<<dim3(DFF / 32, D_MODEL / 32), dim3(32, 8), 0, stream>>>(W1, W1T, D_MODEL, DFF);
  transpose_cvt<<<dim3(D_MODEL / 32, DFF / 32), dim3(32, 8), 0, stream>>>(W2, W2T, DFF, D_MODEL);

  // 2) QKV projections (WMMA), writing attention-friendly layouts
  dim3 gp(D_MODEL / 256, MROWS / 64);
  gemm_bf16_wmma<2><<<gp, 128, 0, stream>>>(xb, WqT, bq, qbuf, MROWS, D_MODEL, D_MODEL);
  gemm_bf16_wmma<2><<<gp, 128, 0, stream>>>(xb, WkT, bk, kbuf, MROWS, D_MODEL, D_MODEL);
  gemm_bf16_wmma<3><<<gp, 128, 0, stream>>>(xb, WvT, bv, vtb,  MROWS, D_MODEL, D_MODEL);

  // 3) masked flash attention (WMMA for QK^T and PV)
  attn_flash_wmma<<<dim3(SEQ / 64, NHEAD, BATCH), 128, 0, stream>>>(qbuf, kbuf, vtb, mask, ctxb);

  // 4) output projection + residual + LN1
  gemm_bf16_wmma<0><<<gp, 128, 0, stream>>>(ctxb, WoT, bo, aout, MROWS, D_MODEL, D_MODEL);
  ln_residual<<<MROWS, 256, 0, stream>>>(x, aout, g1, be1, y1f, y1b);

  // 5) FFN (relu hidden in bf16) + residual + LN2
  gemm_bf16_wmma<1><<<dim3(DFF / 256, MROWS / 64), 128, 0, stream>>>(y1b, W1T, b1f, hb, MROWS, DFF, D_MODEL);
  gemm_bf16_wmma<0><<<gp, 128, 0, stream>>>(hb, W2T, b2f, ffw, MROWS, D_MODEL, DFF);
  ln_residual<<<MROWS, 256, 0, stream>>>(y1f, ffw, g2, be2, out, (__bf16*)nullptr);
}
